// GroupedQueryAttention_36618891166200
// MI455X (gfx1250) — matrix operand order repaired and checked by reading, compiled, not run on hardware
//
#include <hip/hip_runtime.h>
#include <hip/hip_bf16.h>

typedef __bf16 bf16;
typedef __attribute__((ext_vector_type(16))) __bf16 v16bf;
typedef __attribute__((ext_vector_type(8)))  __bf16 v8bf;
typedef __attribute__((ext_vector_type(4)))  __bf16 v4bf;
typedef __attribute__((ext_vector_type(8)))  float  v8f;
typedef __attribute__((ext_vector_type(4)))  int    i32x4;

#define B_   2
#define T_   2048
#define E_   2048
#define H_   32
#define HKV_ 8
#define G_   4
#define D_   64
#define M_   (B_ * T_)   // 4096 rows for all GEMMs

// ---------------------------------------------------------------------------
// CDNA5 async global->LDS staging (ASYNCcnt path). The builtin takes
// (int4 addrspace(1)*, int4 addrspace(3)*, imm offset, imm cpol).
// Guarded so the file still compiles on toolchains without it.
// ---------------------------------------------------------------------------
#if defined(__has_builtin)
#if __has_builtin(__builtin_amdgcn_global_load_async_to_lds_b128)
#define HAVE_ASYNC_LDS 1
#endif
#endif
#ifndef HAVE_ASYNC_LDS
#define HAVE_ASYNC_LDS 0
#endif

#if HAVE_ASYNC_LDS
typedef __attribute__((address_space(1))) i32x4 gi32x4;
typedef __attribute__((address_space(3))) i32x4 li32x4;
#endif

__device__ __forceinline__ void stage16(const bf16* g, bf16* l) {
#if HAVE_ASYNC_LDS
  __builtin_amdgcn_global_load_async_to_lds_b128(
      (gi32x4*)(unsigned long long)g,
      (li32x4*)(unsigned)(unsigned long long)l, 0, 0);
#else
  *(v8bf*)l = *(const v8bf*)g;
#endif
}

__device__ __forceinline__ void wait_async_stage() {
#if HAVE_ASYNC_LDS
#if __has_builtin(__builtin_amdgcn_s_wait_asynccnt)
  __builtin_amdgcn_s_wait_asynccnt(0);
#else
  asm volatile("s_wait_asynccnt 0x0" ::: "memory");
#endif
#endif
}

// ---------------------------------------------------------------------------
// Fragment loaders (CDNA5 ISA); A and B share a contiguous K order, not the ISA table order:
// A 16x32 bf16: lane holds row M=lane&15; half 0 holds K=0..15,
//               half 1 holds K=16..31  -> one contiguous 32B chunk at +16h.
// B 32x16 bf16: lane holds col N=lane&15; half 0 holds K=0..15, half 1
//               K=16..31 -> one contiguous 32B chunk when source is N-major.
// C/D 16x16 f32: vgpr j holds row M=j+8*(lane>=16), col N=lane&15.
// ---------------------------------------------------------------------------
__device__ __forceinline__ v16bf load_a_frag(const bf16* p) {
  return *(const v16bf*)p;
}
__device__ __forceinline__ v16bf load_b_frag(const bf16* p) {
  return *(const v16bf*)p;   // 32 contiguous bytes (K-contiguous, N-major src)
}
__device__ __forceinline__ v8f wmma_bf16(v16bf a, v16bf b, v8f c) {
  return __builtin_amdgcn_wmma_f32_16x16x32_bf16(false, a, false, b,
                                                 (short)0, c, false, false);
}

// ---------------------------------------------------------------------------
// f32 -> bf16 converter, 4 elements / thread
// ---------------------------------------------------------------------------
__global__ void cvt_f32_bf16(const float* __restrict__ s, bf16* __restrict__ d,
                             int n4) {
  int i = blockIdx.x * blockDim.x + threadIdx.x;
  if (i < n4) {
    float4 f = ((const float4*)s)[i];
    v4bf o;
    o[0] = (bf16)f.x; o[1] = (bf16)f.y; o[2] = (bf16)f.z; o[3] = (bf16)f.w;
    ((v4bf*)d)[i] = o;
  }
}

// ---------------------------------------------------------------------------
// Tiled bf16 WMMA GEMM: C[M x N] = A[M x K] * B[K x N]
// Block tile 128x128, 8 waves in a 4(row)x2(col) grid; each wave owns a
// 32x64 sub-tile (2 row frags x 4 col frags = 8 WMMAs per k-step).
// mode 0: scatter bf16 into (b, head, t, d) layout (HH heads). mode 1: f32 rows.
// ---------------------------------------------------------------------------
__global__ __launch_bounds__(256)
void gemm_bf16_wmma(const bf16* __restrict__ A, const bf16* __restrict__ Bm,
                    void* __restrict__ Out, int N, int K, int mode, int HH) {
  __shared__ alignas(128) bf16 As[128 * 32];   // row-major [row][k]
  __shared__ alignas(128) bf16 Bt[128 * 32];   // transposed [n][k]

  const int tid  = threadIdx.x;
  const int lane = tid & 31;
  const int wave = tid >> 5;
  const int wr   = wave & 3;        // wave row group (4)
  const int wc   = wave >> 2;       // wave col group (2)
  const int half = lane >> 4;
  const int ln   = lane & 15;
  const int rb0  = blockIdx.y * 128;
  const int cb0  = blockIdx.x * 128;

  v8f acc[2][4];
#pragma unroll
  for (int r = 0; r < 2; ++r)
#pragma unroll
    for (int c = 0; c < 4; ++c) acc[r][c] = v8f{};

  const int kSteps = K >> 5;
  const int arow = tid >> 1;        // 0..127  (A-tile row)
  const int aoff = (tid & 1) * 16;  // 0 or 16 (A-tile k half)
  const int brow = tid >> 3;        // 0..31   (B-tile k row)
  const int bcol = (tid & 7) * 16;  // 0..112  (B-tile n offset)

  for (int kt = 0; kt < kSteps; ++kt) {
    const int k0 = kt * 32;
    // ---- stage A tile (async DMA to LDS when available) ----
    {
      const bf16* ag = A + (size_t)(rb0 + arow) * K + k0 + aoff;
      bf16* as = As + arow * 32 + aoff;
      stage16(ag, as);
      stage16(ag + 8, as + 8);
    }
    // ---- stage B tile transposed into [n][k] (needs VGPR round-trip) ----
    {
      const bf16* bg = Bm + (size_t)(k0 + brow) * N + cb0 + bcol;
      v8bf b0 = *(const v8bf*)(bg);
      v8bf b1 = *(const v8bf*)(bg + 8);
#pragma unroll
      for (int e = 0; e < 8; ++e) {
        Bt[(bcol + e) * 32 + brow]     = b0[e];
        Bt[(bcol + 8 + e) * 32 + brow] = b1[e];
      }
    }
    if (kt + 1 < kSteps) {  // prefetch next tiles
      __builtin_prefetch(A + (size_t)(rb0 + arow) * K + k0 + 32 + aoff, 0, 1);
      __builtin_prefetch(Bm + (size_t)(k0 + 32 + brow) * N + cb0 + bcol, 0, 1);
    }
    wait_async_stage();
    __syncthreads();

    // ---- compute: 2 A frags x 4 B frags = 8 WMMAs ----
    v16bf af[2];
#pragma unroll
    for (int r = 0; r < 2; ++r)
      af[r] = load_a_frag(As + (wr * 32 + r * 16 + ln) * 32 + 16 * half);
#pragma unroll
    for (int c = 0; c < 4; ++c) {
      v16bf bfv = load_b_frag(Bt + (wc * 64 + c * 16 + ln) * 32 + 16 * half);
#pragma unroll
      for (int r = 0; r < 2; ++r) acc[r][c] = wmma_bf16(af[r], bfv, acc[r][c]);
    }
    __syncthreads();
  }

  // ---- epilogue ----
#pragma unroll
  for (int r = 0; r < 2; ++r) {
#pragma unroll
    for (int c = 0; c < 4; ++c) {
#pragma unroll
      for (int j = 0; j < 8; ++j) {
        const int row = rb0 + wr * 32 + r * 16 + j + 8 * half;
        const int col = cb0 + wc * 64 + c * 16 + ln;
        const float v = acc[r][c][j];
        if (mode == 0) {
          const int b = row >> 11, t = row & (T_ - 1);
          const int head = col >> 6, dd = col & 63;
          ((bf16*)Out)[(((size_t)b * HH + head) * T_ + t) * D_ + dd] = (bf16)v;
        } else {
          ((float*)Out)[(size_t)row * N + col] = v;
        }
      }
    }
  }
}

// ---------------------------------------------------------------------------
// RoPE (in place, f32 math on bf16 storage), layout (b, head, t, d)
// thread handles rotation pair (d, d+32). total = B*heads*T*32, T*32 = 2^16.
// ---------------------------------------------------------------------------
__global__ void rope_kernel(bf16* __restrict__ q, int total) {
  const int idx = blockIdx.x * blockDim.x + threadIdx.x;
  if (idx >= total) return;
  const int i  = idx & 31;
  const int t  = (idx >> 5) & (T_ - 1);
  const int bh = idx >> 16;
  const size_t base = (size_t)bh * T_ * D_ + (size_t)t * D_;
  const float inv = __powf(10000.0f, -(float)i / 32.0f);
  const float ang = (float)t * inv;
  const float c = __cosf(ang), s = __sinf(ang);
  const float x1 = (float)q[base + i];
  const float x2 = (float)q[base + i + 32];
  q[base + i]      = (bf16)(x1 * c - x2 * s);
  q[base + i + 32] = (bf16)(x2 * c + x1 * s);
}

// ---------------------------------------------------------------------------
// Causal GQA flash attention. Block = (16-query tile, one KV head); the 4
// waves are the 4 query heads of the group, sharing LDS K/V tiles.
// Q,K stored (b, head, t, d) bf16; output Y stored (b, t, h*64+d) bf16.
// ---------------------------------------------------------------------------
__global__ __launch_bounds__(128)
void flash_attn(const bf16* __restrict__ Q, const bf16* __restrict__ Kb,
                const bf16* __restrict__ Vb, bf16* __restrict__ Y) {
  __shared__ alignas(128) bf16 Ks[32 * 64];       // [key][d]
  __shared__ alignas(128) bf16 Vt[64 * 32];       // [d][key]
  __shared__ alignas(128) bf16 Ps[4][16 * 32];    // per-wave P tile [m][key]

  const int tid  = threadIdx.x;
  const int lane = tid & 31;
  const int wave = tid >> 5;
  const int half = lane >> 4;
  const int ln   = lane & 15;
  const int t0   = blockIdx.x * 16;
  const int bk   = blockIdx.y;            // b*HKV + kh
  const int b    = bk >> 3;
  const int kh   = bk & (HKV_ - 1);
  const int h    = kh * G_ + wave;

  // Q fragments (row = t0+ln, D split into two 16x32 A fragments)
  const bf16* qptr = Q + (((size_t)b * H_ + h) * T_ + t0) * D_;
  v16bf aq0, aq1;
  {
    const bf16* p = qptr + (size_t)ln * D_ + 16 * half;
    aq0 = load_a_frag(p);
    aq1 = load_a_frag(p + 32);
  }

  v8f y0 = v8f{}, y1 = v8f{}, y2 = v8f{}, y3 = v8f{};
  float mrow[8], lrow[8];
#pragma unroll
  for (int j = 0; j < 8; ++j) { mrow[j] = -3.0e38f; lrow[j] = 0.0f; }

  const bf16* kbase = Kb + (size_t)bk * T_ * D_;
  const bf16* vbase = Vb + (size_t)bk * T_ * D_;
  const int nkb  = (t0 + 16 + 31) >> 5;   // number of 32-key blocks (causal)
  const int srow = tid >> 2;              // staging key row 0..31
  const int scol = (tid & 3) * 16;        // staging d offset

  for (int kb = 0; kb < nkb; ++kb) {
    const int k0 = kb * 32;
    // ---- stage K (async DMA) and V (transposed via VGPRs) ----
    {
      const bf16* kg = kbase + (size_t)(k0 + srow) * D_ + scol;
      bf16* ks = Ks + srow * 64 + scol;
      stage16(kg, ks);
      stage16(kg + 8, ks + 8);
      const bf16* vg = vbase + (size_t)(k0 + srow) * D_ + scol;
      v8bf va = *(const v8bf*)vg;
      v8bf vc = *(const v8bf*)(vg + 8);
#pragma unroll
      for (int e = 0; e < 8; ++e) {
        Vt[(scol + e) * 32 + srow]     = va[e];
        Vt[(scol + 8 + e) * 32 + srow] = vc[e];
      }
    }
    if (kb + 1 < nkb) {
      __builtin_prefetch(kbase + (size_t)(k0 + 32 + srow) * D_ + scol, 0, 1);
      __builtin_prefetch(vbase + (size_t)(k0 + 32 + srow) * D_ + scol, 0, 1);
    }
    wait_async_stage();
    __syncthreads();

    // ---- scores: S(16q x 32k) = Q(16x64) K^T, 4 WMMAs ----
    v8f s0 = v8f{}, s1 = v8f{};
    {
      const bf16* p0 = Ks + (size_t)ln * 64 + 16 * half;          // keys 0..15
      const bf16* p1 = Ks + (size_t)(16 + ln) * 64 + 16 * half;   // keys 16..31
      s0 = wmma_bf16(aq0, load_b_frag(p0), s0);
      s0 = wmma_bf16(aq1, load_b_frag(p0 + 32), s0);
      s1 = wmma_bf16(aq0, load_b_frag(p1), s1);
      s1 = wmma_bf16(aq1, load_b_frag(p1 + 32), s1);
    }

    // ---- online softmax (row j+8*half lives only in this 16-lane half) ----
#pragma unroll
    for (int j = 0; j < 8; ++j) {
      const int t = t0 + j + 8 * half;
      float v0 = s0[j] * 0.125f;
      float v1 = s1[j] * 0.125f;
      if (k0 + ln > t)      v0 = -3.0e38f;
      if (k0 + 16 + ln > t) v1 = -3.0e38f;
      float mb = fmaxf(v0, v1);
#pragma unroll
      for (int off = 8; off; off >>= 1) mb = fmaxf(mb, __shfl_xor(mb, off, 32));
      const float mnew = fmaxf(mrow[j], mb);
      const float corr = __expf(mrow[j] - mnew);
      const float p0 = __expf(v0 - mnew);
      const float p1 = __expf(v1 - mnew);
      float rs = p0 + p1;
#pragma unroll
      for (int off = 8; off; off >>= 1) rs += __shfl_xor(rs, off, 32);
      lrow[j] = lrow[j] * corr + rs;
      mrow[j] = mnew;
      y0[j] *= corr; y1[j] *= corr; y2[j] *= corr; y3[j] *= corr;
      Ps[wave][(j + 8 * half) * 32 + ln]      = (bf16)p0;
      Ps[wave][(j + 8 * half) * 32 + 16 + ln] = (bf16)p1;
    }

    // ---- PV: Y(16 x 64) += P(16x32) V(32x64), 4 WMMAs ----
    {
      v16bf ap = load_a_frag(&Ps[wave][(size_t)ln * 32 + 16 * half]);
      y0 = wmma_bf16(ap, load_b_frag(Vt + (size_t)(0 + ln) * 32 + 16 * half), y0);
      y1 = wmma_bf16(ap, load_b_frag(Vt + (size_t)(16 + ln) * 32 + 16 * half), y1);
      y2 = wmma_bf16(ap, load_b_frag(Vt + (size_t)(32 + ln) * 32 + 16 * half), y2);
      y3 = wmma_bf16(ap, load_b_frag(Vt + (size_t)(48 + ln) * 32 + 16 * half), y3);
    }
    __syncthreads();
  }

  // ---- finalize: divide by row sums, write (b, t, h*64+d) bf16 ----
  bf16* yout = Y + ((size_t)b * T_ + t0) * (H_ * D_) + (size_t)h * D_;
#pragma unroll
  for (int j = 0; j < 8; ++j) {
    const float inv = 1.0f / lrow[j];
    bf16* yp = yout + (size_t)(j + 8 * half) * (H_ * D_);
    yp[0 * 16 + ln] = (bf16)(y0[j] * inv);
    yp[1 * 16 + ln] = (bf16)(y1[j] * inv);
    yp[2 * 16 + ln] = (bf16)(y2[j] * inv);
    yp[3 * 16 + ln] = (bf16)(y3[j] * inv);
  }
}

// ---------------------------------------------------------------------------
extern "C" void kernel_launch(void* const* d_in, const int* in_sizes, int n_in,
                              void* d_out, int out_size, void* d_ws,
                              size_t ws_size, hipStream_t stream) {
  const float* x  = (const float*)d_in[0];
  const float* Wq = (const float*)d_in[1];
  const float* Wk = (const float*)d_in[2];
  const float* Wv = (const float*)d_in[3];
  const float* Wo = (const float*)d_in[4];

  char* ws = (char*)d_ws;
  size_t off = 0;
  auto alloc = [&](size_t bytes) -> void* {
    void* p = ws + off;
    off += (bytes + 255) & ~(size_t)255;
    return p;
  };
  bf16* xb  = (bf16*)alloc((size_t)M_ * E_ * 2);
  bf16* Wqb = (bf16*)alloc((size_t)E_ * H_ * D_ * 2);
  bf16* Wkb = (bf16*)alloc((size_t)E_ * HKV_ * D_ * 2);
  bf16* Wvb = (bf16*)alloc((size_t)E_ * HKV_ * D_ * 2);
  bf16* Wob = (bf16*)alloc((size_t)E_ * E_ * 2);
  bf16* qb  = (bf16*)alloc((size_t)B_ * H_ * T_ * D_ * 2);
  bf16* kb  = (bf16*)alloc((size_t)B_ * HKV_ * T_ * D_ * 2);
  bf16* vb  = (bf16*)alloc((size_t)B_ * HKV_ * T_ * D_ * 2);
  bf16* yb  = (bf16*)alloc((size_t)M_ * E_ * 2);

  auto cvt = [&](const float* s, bf16* d, size_t n) {
    int n4 = (int)(n / 4);
    cvt_f32_bf16<<<dim3((n4 + 255) / 256), dim3(256), 0, stream>>>(s, d, n4);
  };
  cvt(x,  xb,  (size_t)M_ * E_);
  cvt(Wq, Wqb, (size_t)E_ * H_ * D_);
  cvt(Wk, Wkb, (size_t)E_ * HKV_ * D_);
  cvt(Wv, Wvb, (size_t)E_ * HKV_ * D_);
  cvt(Wo, Wob, (size_t)E_ * E_);

  // q/k/v projections (scatter epilogue into (b, head, t, d))
  gemm_bf16_wmma<<<dim3((H_ * D_) / 128, M_ / 128), 256, 0, stream>>>(
      xb, Wqb, qb, H_ * D_, E_, 0, H_);
  gemm_bf16_wmma<<<dim3((HKV_ * D_) / 128, M_ / 128), 256, 0, stream>>>(
      xb, Wkb, kb, HKV_ * D_, E_, 0, HKV_);
  gemm_bf16_wmma<<<dim3((HKV_ * D_) / 128, M_ / 128), 256, 0, stream>>>(
      xb, Wvb, vb, HKV_ * D_, E_, 0, HKV_);

  // RoPE on q and k
  {
    int tq = B_ * H_ * T_ * 32;
    rope_kernel<<<dim3((tq + 255) / 256), dim3(256), 0, stream>>>(qb, tq);
    int tk = B_ * HKV_ * T_ * 32;
    rope_kernel<<<dim3((tk + 255) / 256), dim3(256), 0, stream>>>(kb, tk);
  }

  // attention
  flash_attn<<<dim3(T_ / 16, B_ * HKV_), 128, 0, stream>>>(qb, kb, vb, yb);

  // output projection -> f32 d_out
  gemm_bf16_wmma<<<dim3(E_ / 128, M_ / 128), 256, 0, stream>>>(
      yb, Wob, d_out, E_, E_, 1, 0);
}
